// FrameTransformerEncoder_66202625900954
// MI455X (gfx1250) — matrix operand order repaired and checked by reading, not compiled, not run on hardware
//
#include <hip/hip_runtime.h>
#include <hip/hip_bf16.h>

typedef __attribute__((ext_vector_type(16))) __bf16 v16bf;
typedef __attribute__((ext_vector_type(8)))  float  v8f;
typedef unsigned short u16;
typedef unsigned int   u32;

// ---------- constants ----------
#define BB    2
#define CIN   2
#define CC    4
#define FEAT  512
#define FRAMES 1024
#define HEADS 16
#define HDIM  32
#define EXPF  4
#define GG    (BB*CC)                 // 8 batched (b,c) slices
#define GFN   ((size_t)GG*FEAT*FRAMES)

// round-to-nearest-even f32 -> bf16
__device__ __forceinline__ u16 f2bf(float f) {
    u32 u = __float_as_uint(f);
    u32 r = u + 0x7FFFu + ((u >> 16) & 1u);
    return (u16)(r >> 16);
}

union FragBF {
    v16bf v;
    uint4 q[2];
};

__device__ __forceinline__ v8f wmma_bf16(v16bf a, v16bf b, v8f c) {
    return __builtin_amdgcn_wmma_f32_16x16x32_bf16(false, a, false, b, (short)0, c, false, false);
}

// generic->LDS byte offset: low 32 bits of the flat address are the LDS offset
__device__ __forceinline__ unsigned lds_off(const void* p) {
    return (unsigned)(uintptr_t)p;
}

// CDNA5 async global->LDS copy, 16B per active lane (tracked by ASYNCcnt)
__device__ __forceinline__ void async_g2l_b128(unsigned loff, const void* gptr) {
    asm volatile("global_load_async_to_lds_b128 %0, %1, off"
                 :: "v"(loff), "v"((unsigned long long)(uintptr_t)gptr)
                 : "memory");
}
__device__ __forceinline__ void wait_async0() {
    asm volatile("s_wait_asynccnt 0" ::: "memory");
}

// ---------- 1) embed: x0[b,co,f,w] = sum_ci ew[co,ci]*x[b,ci,f,w] + eb[co] ----------
__global__ void embed_kernel(const float* __restrict__ x, const float* __restrict__ ew,
                             const float* __restrict__ eb, float* __restrict__ x0) {
    size_t idx = (size_t)blockIdx.x * blockDim.x + threadIdx.x;
    if (idx >= GFN) return;
    int w  = (int)(idx & 1023);
    int f  = (int)((idx >> 10) & 511);
    int co = (int)((idx >> 19) & 3);
    int b  = (int)(idx >> 21);
    float acc = eb[co];
#pragma unroll
    for (int ci = 0; ci < CIN; ++ci)
        acc += ew[co * CIN + ci] * x[((size_t)(b * CIN + ci) * FEAT + f) * FRAMES + w];
    x0[idx] = acc;
}

// ---------- 2) frame norm over FEAT per (g, frame); emit bf16 TRANSPOSED (g, frame, feat) ----------
__global__ void framenorm_kernel(const float* __restrict__ xin, const float* __restrict__ nw,
                                 const float* __restrict__ nb, u16* __restrict__ outT) {
    int t = blockIdx.x * blockDim.x + threadIdx.x;     // (g, w)
    if (t >= GG * FRAMES) return;
    int w = t & 1023;
    int g = t >> 10;
    int c = g & 3;
    const float* p = xin + (size_t)g * FEAT * FRAMES + w;
    float s = 0.f, ss = 0.f;
    for (int f = 0; f < FEAT; ++f) {
        float v = p[(size_t)f * FRAMES];
        s += v; ss += v * v;
    }
    float mu  = s * (1.0f / FEAT);
    float var = ss * (1.0f / FEAT) - mu * mu;
    float rstd = rsqrtf(var + 1e-8f);
    u16* o = outT + ((size_t)g * FRAMES + w) * FEAT;
    for (int f = 0; f < FEAT; ++f) {
        float v = (p[(size_t)f * FRAMES] - mu) * rstd * nw[c * FEAT + f] + nb[c * FEAT + f];
        o[f] = f2bf(v);
    }
}

// ---------- 3) generic f32 -> bf16 convert (weights) ----------
__global__ void cvt_bf16_kernel(const float* __restrict__ in, u16* __restrict__ out, size_t n) {
    size_t i = (size_t)blockIdx.x * blockDim.x + threadIdx.x;
    if (i < n) out[i] = f2bf(in[i]);
}

// ---------- 4) batched bf16 WMMA GEMM: Y[g] = W[g%CC] (MxK) * X[g] (N x K, transposed) ----------
// block tile 128x64, 8 waves, wave tile 32x64, K step 32, double-buffered async LDS staging.
__global__ __launch_bounds__(256) void gemm_bf16_kernel(
        const u16* __restrict__ W, const u16* __restrict__ Xt,
        const float* __restrict__ addsrc, float* __restrict__ outf,
        u16* __restrict__ outbfT, int M, int N, int K, int relu2) {
    __shared__ u16 As[2][128][40];   // A tile (M x K), rows 16B-aligned (80B stride)
    __shared__ u16 Bs[2][64][40];    // B tile (N x K)
    const int g    = blockIdx.z;
    const int c    = g & (CC - 1);
    const u16* Wg  = W  + (size_t)c * M * K;
    const u16* Xg  = Xt + (size_t)g * N * K;
    const int m0   = blockIdx.y * 128;
    const int n0   = blockIdx.x * 64;
    const int tid  = threadIdx.x;
    const int lane = tid & 31;
    const int wv   = tid >> 5;
    const int ln15 = lane & 15;
    const int hf   = lane >> 4;

    v8f acc[2][4];
#pragma unroll
    for (int mi = 0; mi < 2; ++mi)
#pragma unroll
        for (int ni = 0; ni < 4; ++ni) acc[mi][ni] = (v8f){};

    // async stagers: contiguous 16B per lane
    auto stageA = [&](int buf, int k0) {
#pragma unroll
        for (int u = 0; u < 2; ++u) {
            int e = tid + u * 256;                 // 0..511 -> 128 rows x 4 parts
            int m = e >> 2, part = e & 3;
            async_g2l_b128(lds_off(&As[buf][m][part * 8]),
                           Wg + (size_t)(m0 + m) * K + k0 + part * 8);
        }
    };
    auto stageB = [&](int buf, int k0) {
        int n = tid >> 2, part = tid & 3;          // 64 rows x 4 parts
        async_g2l_b128(lds_off(&Bs[buf][n][part * 8]),
                       Xg + (size_t)(n0 + n) * K + k0 + part * 8);
    };

    stageA(0, 0);
    stageB(0, 0);
    const int nk = K / 32;
    for (int ki = 0; ki < nk; ++ki) {
        const int buf = ki & 1;
        wait_async0();
        __syncthreads();                           // tile `buf` visible to all waves
        if (ki + 1 < nk) {                         // prefetch next tile into other buffer
            stageA(buf ^ 1, (ki + 1) * 32);
            stageB(buf ^ 1, (ki + 1) * 32);
        }
        const int kb = hf * 8;
        FragBF af[2], bfm[4];
#pragma unroll
        for (int mi = 0; mi < 2; ++mi) {
            const u16* row = &As[buf][wv * 32 + mi * 16 + ln15][0];
            af[mi].q[0] = *(const uint4*)(row + kb);
            af[mi].q[1] = *(const uint4*)(row + kb + 16);
        }
        const int ko = hf * 8;
#pragma unroll
        for (int ni = 0; ni < 4; ++ni) {
            const u16* row = &Bs[buf][ni * 16 + ln15][0];
            bfm[ni].q[0] = *(const uint4*)(row + ko);
            bfm[ni].q[1] = *(const uint4*)(row + ko + 16);
        }
#pragma unroll
        for (int mi = 0; mi < 2; ++mi)
#pragma unroll
            for (int ni = 0; ni < 4; ++ni)
                acc[mi][ni] = wmma_bf16(af[mi].v, bfm[ni].v, acc[mi][ni]);
    }

    // epilogue: C layout row = r + 8*hf (per 16x16 tile)
#pragma unroll
    for (int mi = 0; mi < 2; ++mi)
#pragma unroll
        for (int ni = 0; ni < 4; ++ni)
#pragma unroll
            for (int r = 0; r < 8; ++r) {
                int row = m0 + wv * 32 + mi * 16 + r + 8 * hf;
                int col = n0 + ni * 16 + ln15;
                size_t idx = (size_t)g * M * N + (size_t)row * N + col;
                float v = acc[mi][ni][r];
                if (relu2) { v = fmaxf(v, 0.f); v = v * v; }
                if (addsrc) v += addsrc[idx];
                if (outf)   outf[idx] = v;
                if (outbfT) outbfT[(size_t)g * N * M + (size_t)col * M + row] = f2bf(v);
            }
}

// ---------- 5) conv1x3 (channel mix, 3-tap over frames) + bias + optional RoPE ----------
// q/k (do_rope=1): pack [g][h][frame][HDIM];  v (do_rope=0): pack [g][h][HDIM][frame]
__global__ void conv_rope_pack_kernel(const float* __restrict__ in, const float* __restrict__ cw,
                                      const float* __restrict__ cb, u16* __restrict__ outp,
                                      int do_rope) {
    int idx = blockIdx.x * blockDim.x + threadIdx.x;   // 2*4*16*1024*16 = 2^21
    if (idx >= (1 << 21)) return;
    int i  = idx & 15;           // d-pair index within head
    int w  = (idx >> 4) & 1023;  // frame
    int h  = (idx >> 14) & 15;   // head
    int co = (idx >> 18) & 3;    // channel out
    int b  = idx >> 20;
    int f0 = h * HDIM + 2 * i;
    float y0 = cb[co], y1 = cb[co];
#pragma unroll
    for (int ci = 0; ci < CC; ++ci) {
        const float* base = in + (size_t)(b * CC + ci) * FEAT * FRAMES;
#pragma unroll
        for (int t = 0; t < 3; ++t) {
            int wp = w + t - 1;
            if (wp < 0 || wp > FRAMES - 1) continue;
            float wt = cw[(co * CC + ci) * 3 + t];
            y0 += wt * base[(size_t)f0 * FRAMES + wp];
            y1 += wt * base[(size_t)(f0 + 1) * FRAMES + wp];
        }
    }
    size_t gh = (size_t)((b * CC + co) * HEADS + h);
    if (do_rope) {
        float inv = powf(10000.0f, -(float)(2 * i) / (float)HDIM);
        float ang = (float)w * inv;
        float cs = cosf(ang), sn = sinf(ang);
        float r0 = y0 * cs - y1 * sn;
        float r1 = y1 * cs + y0 * sn;
        size_t o = (gh * FRAMES + w) * HDIM + 2 * i;
        outp[o]     = f2bf(r0);
        outp[o + 1] = f2bf(r1);
    } else {
        size_t o = (gh * HDIM + 2 * i) * FRAMES + w;   // d-major for async V staging
        outp[o]          = f2bf(y0);
        outp[o + FRAMES] = f2bf(y1);
    }
}

// ---------- 6) flash attention per (g, head); 4 waves x 32 query rows ----------
// Q,K: [g][h][frame][HDIM]; V: [g][h][HDIM][frame]; out: bf16 TRANSPOSED (g, frame, feat)
__global__ __launch_bounds__(128) void attention_kernel(
        const u16* __restrict__ Qp, const u16* __restrict__ Kp,
        const u16* __restrict__ Vp, u16* __restrict__ ObT) {
    __shared__ u16 Vt[HDIM][40];        // V tile: [d][j]
    __shared__ u16 Ps[4][32][40];       // per-wave P staging (A-layout source)
    const int qb = blockIdx.x;          // 0..7
    const int h  = blockIdx.y;          // 0..15
    const int g  = blockIdx.z;          // 0..7
    const int tid = threadIdx.x, lane = tid & 31, wv = tid >> 5;
    const int hf = lane >> 4, ln15 = lane & 15;
    const int q0 = qb * 128 + wv * 32;
    const u16* Qg = Qp + (size_t)(g * HEADS + h) * FRAMES * HDIM;
    const u16* Kg = Kp + (size_t)(g * HEADS + h) * FRAMES * HDIM;
    const u16* Vg = Vp + (size_t)(g * HEADS + h) * HDIM * FRAMES;   // d-major

    // Q fragments once (A-matrix, K = HDIM = 32 exactly)
    const int kb = hf * 8;
    FragBF qf[2];
#pragma unroll
    for (int mi = 0; mi < 2; ++mi) {
        const u16* row = Qg + (size_t)(q0 + mi * 16 + ln15) * HDIM;
        qf[mi].q[0] = *(const uint4*)(row + kb);
        qf[mi].q[1] = *(const uint4*)(row + kb + 16);
    }

    v8f accO[2][2];
#pragma unroll
    for (int mi = 0; mi < 2; ++mi) { accO[mi][0] = (v8f){}; accO[mi][1] = (v8f){}; }
    float mst[2][8], lst[2][8];
#pragma unroll
    for (int mi = 0; mi < 2; ++mi)
#pragma unroll
        for (int r = 0; r < 8; ++r) { mst[mi][r] = -3.0e38f; lst[mi][r] = 0.f; }

    const float sscale = 0.044194173824159216f;   // 1/sqrt(FEAT=512)

    for (int jb = 0; jb < FRAMES; jb += 32) {
        __syncthreads();                          // previous iter's V consumers done
        {   // async stage V tile: 32 d-rows x 64B, one 16B op per thread
            int d = tid >> 2, part = tid & 3;
            async_g2l_b128(lds_off(&Vt[d][part * 8]),
                           Vg + (size_t)d * FRAMES + jb + part * 8);
        }
        wait_async0();
        __syncthreads();

        // K fragments direct from global (B-matrix: B[d][j] = K[j][d])
        FragBF kf[2];
        const int kd = hf * 8;
#pragma unroll
        for (int nj = 0; nj < 2; ++nj) {
            const u16* row = Kg + (size_t)(jb + nj * 16 + ln15) * HDIM + kd;
            kf[nj].q[0] = *(const uint4*)(row);
            kf[nj].q[1] = *(const uint4*)(row + 16);
        }
        v8f accS[2][2];
#pragma unroll
        for (int mi = 0; mi < 2; ++mi) {
            accS[mi][0] = (v8f){}; accS[mi][1] = (v8f){};
#pragma unroll
            for (int nj = 0; nj < 2; ++nj)
                accS[mi][nj] = wmma_bf16(qf[mi].v, kf[nj].v, accS[mi][nj]);
        }

        // online softmax (each output row lives in one 16-lane half)
#pragma unroll
        for (int mi = 0; mi < 2; ++mi)
#pragma unroll
            for (int r = 0; r < 8; ++r) {
                float s0 = accS[mi][0][r] * sscale;
                float s1 = accS[mi][1][r] * sscale;
                float mx = fmaxf(s0, s1);
#pragma unroll
                for (int o = 1; o < 16; o <<= 1) mx = fmaxf(mx, __shfl_xor(mx, o, 32));
                float mnew  = fmaxf(mst[mi][r], mx);
                float alpha = __expf(mst[mi][r] - mnew);
                float p0 = __expf(s0 - mnew);
                float p1 = __expf(s1 - mnew);
                float rs = p0 + p1;
#pragma unroll
                for (int o = 1; o < 16; o <<= 1) rs += __shfl_xor(rs, o, 32);
                lst[mi][r] = lst[mi][r] * alpha + rs;
                mst[mi][r] = mnew;
                accO[mi][0][r] *= alpha;
                accO[mi][1][r] *= alpha;
                int prow = mi * 16 + r + 8 * hf;
                Ps[wv][prow][ln15]      = f2bf(p0);
                Ps[wv][prow][16 + ln15] = f2bf(p1);
            }

        // P * V  (same-wave LDS ops are in-order, store->load is safe)
        FragBF pf[2], vf[2];
#pragma unroll
        for (int mi = 0; mi < 2; ++mi) {
            const u16* row = &Ps[wv][mi * 16 + ln15][0];
            pf[mi].q[0] = *(const uint4*)(row + kb);
            pf[mi].q[1] = *(const uint4*)(row + kb + 16);
        }
        const int ko = hf * 8;
#pragma unroll
        for (int nt = 0; nt < 2; ++nt) {
            const u16* row = &Vt[nt * 16 + ln15][0];
            vf[nt].q[0] = *(const uint4*)(row + ko);
            vf[nt].q[1] = *(const uint4*)(row + ko + 16);
        }
#pragma unroll
        for (int mi = 0; mi < 2; ++mi)
#pragma unroll
            for (int nt = 0; nt < 2; ++nt)
                accO[mi][nt] = wmma_bf16(pf[mi].v, vf[nt].v, accO[mi][nt]);
    }

    // write O as bf16, transposed (g, frame, feat); feat = h*32 + d
#pragma unroll
    for (int mi = 0; mi < 2; ++mi)
#pragma unroll
        for (int nt = 0; nt < 2; ++nt)
#pragma unroll
            for (int r = 0; r < 8; ++r) {
                int qrow = q0 + mi * 16 + r + 8 * hf;
                int d    = nt * 16 + ln15;
                ObT[((size_t)g * FRAMES + qrow) * FEAT + h * HDIM + d] =
                    f2bf(accO[mi][nt][r] / lst[mi][r]);
            }
}

// ============================ host orchestration ============================
extern "C" void kernel_launch(void* const* d_in, const int* in_sizes, int n_in,
                              void* d_out, int out_size, void* d_ws, size_t ws_size,
                              hipStream_t stream) {
    (void)in_sizes; (void)n_in; (void)out_size; (void)ws_size;
    const float* x       = (const float*)d_in[0];
    const float* embed_w = (const float*)d_in[1];
    const float* embed_b = (const float*)d_in[2];
    const float* norm1_w = (const float*)d_in[3];
    const float* norm1_b = (const float*)d_in[4];
    const float* q_pw    = (const float*)d_in[5];
    const float* q_cw    = (const float*)d_in[6];
    const float* q_cb    = (const float*)d_in[7];
    const float* k_pw    = (const float*)d_in[8];
    const float* k_cw    = (const float*)d_in[9];
    const float* k_cb    = (const float*)d_in[10];
    const float* v_pw    = (const float*)d_in[11];
    const float* v_cw    = (const float*)d_in[12];
    const float* v_cb    = (const float*)d_in[13];
    const float* out_pw  = (const float*)d_in[14];
    const float* norm2_w = (const float*)d_in[15];
    const float* norm2_b = (const float*)d_in[16];
    const float* lin1_w  = (const float*)d_in[17];
    const float* lin2_w  = (const float*)d_in[18];
    float* out = (float*)d_out;

    // ---- workspace arena (~160 MB) ----
    char* ws = (char*)d_ws;
    size_t off = 0;
    auto alloc = [&](size_t bytes) -> void* {
        off = (off + 255) & ~(size_t)255;
        void* p = ws + off;
        off += bytes;
        return p;
    };
    float* x0f  = (float*)alloc(GFN * 4);          // running residual accumulator
    u16*   xnb  = (u16*)  alloc(GFN * 2);          // norm1 output bf16 (transposed)
    u16*   wqb  = (u16*)  alloc((size_t)CC*FEAT*FEAT*2);
    u16*   wkb  = (u16*)  alloc((size_t)CC*FEAT*FEAT*2);
    u16*   wvb  = (u16*)  alloc((size_t)CC*FEAT*FEAT*2);
    u16*   wob  = (u16*)  alloc((size_t)CC*FEAT*FEAT*2);
    u16*   wl1b = (u16*)  alloc((size_t)CC*FEAT*EXPF*FEAT*2);
    u16*   wl2b = (u16*)  alloc((size_t)CC*FEAT*FEAT*EXPF*2);
    float* gq   = (float*)alloc(GFN * 4);
    float* gk   = (float*)alloc(GFN * 4);
    float* gv   = (float*)alloc(GFN * 4);
    u16*   Qp   = (u16*)  alloc(GFN * 2);
    u16*   Kp   = (u16*)  alloc(GFN * 2);
    u16*   Vp   = (u16*)  alloc(GFN * 2);
    u16*   ObT  = (u16*)  alloc(GFN * 2);          // attention out bf16 (transposed)
    u16*   xn2b = (u16*)  alloc(GFN * 2);          // norm2 output bf16 (transposed)
    u16*   h1b  = (u16*)  alloc((size_t)GG*FEAT*EXPF*FRAMES*2);  // relu^2 out (transposed)

    // ---- 0) weight conversions to bf16 ----
    auto cvt = [&](const float* src, u16* dst, size_t n) {
        cvt_bf16_kernel<<<(unsigned)((n + 255) / 256), 256, 0, stream>>>(src, dst, n);
    };
    cvt(q_pw,  wqb,  (size_t)CC*FEAT*FEAT);
    cvt(k_pw,  wkb,  (size_t)CC*FEAT*FEAT);
    cvt(v_pw,  wvb,  (size_t)CC*FEAT*FEAT);
    cvt(out_pw,wob,  (size_t)CC*FEAT*FEAT);
    cvt(lin1_w,wl1b, (size_t)CC*FEAT*EXPF*FEAT);
    cvt(lin2_w,wl2b, (size_t)CC*FEAT*FEAT*EXPF);

    // ---- 1) embed + norm1 ----
    embed_kernel<<<(unsigned)(GFN / 256), 256, 0, stream>>>(x, embed_w, embed_b, x0f);
    framenorm_kernel<<<(GG * FRAMES) / 256, 256, 0, stream>>>(x0f, norm1_w, norm1_b, xnb);

    // ---- 2) q/k/v projections (M=512,K=512,N=1024 per g) ----
    dim3 gg(FRAMES / 64, FEAT / 128, GG);
    gemm_bf16_kernel<<<gg, 256, 0, stream>>>(wqb, xnb, nullptr, gq, nullptr, FEAT, FRAMES, FEAT, 0);
    gemm_bf16_kernel<<<gg, 256, 0, stream>>>(wkb, xnb, nullptr, gk, nullptr, FEAT, FRAMES, FEAT, 0);
    gemm_bf16_kernel<<<gg, 256, 0, stream>>>(wvb, xnb, nullptr, gv, nullptr, FEAT, FRAMES, FEAT, 0);

    // ---- 3) conv1x3 + bias (+rope for q,k) + head packing ----
    const unsigned crp_blocks = (1u << 21) / 256;
    conv_rope_pack_kernel<<<crp_blocks, 256, 0, stream>>>(gq, q_cw, q_cb, Qp, 1);
    conv_rope_pack_kernel<<<crp_blocks, 256, 0, stream>>>(gk, k_cw, k_cb, Kp, 1);
    conv_rope_pack_kernel<<<crp_blocks, 256, 0, stream>>>(gv, v_cw, v_cb, Vp, 0);

    // ---- 4) attention -> bf16 transposed ----
    dim3 ag(FRAMES / 128, HEADS, GG);
    attention_kernel<<<ag, 128, 0, stream>>>(Qp, Kp, Vp, ObT);

    // ---- 5) out projection, accumulate into residual in place ----
    gemm_bf16_kernel<<<gg, 256, 0, stream>>>(wob, ObT, x0f, x0f, nullptr, FEAT, FRAMES, FEAT, 0);

    // ---- 6) MLP ----
    framenorm_kernel<<<(GG * FRAMES) / 256, 256, 0, stream>>>(x0f, norm2_w, norm2_b, xn2b);
    dim3 g1(FRAMES / 64, (FEAT * EXPF) / 128, GG);
    gemm_bf16_kernel<<<g1, 256, 0, stream>>>(wl1b, xn2b, nullptr, nullptr, h1b,
                                             FEAT * EXPF, FRAMES, FEAT, 1);        // relu^2 -> bf16 T
    gemm_bf16_kernel<<<gg, 256, 0, stream>>>(wl2b, h1b, x0f, out, nullptr,
                                             FEAT, FRAMES, FEAT * EXPF, 0);        // + residual -> out
}
